// SkeletonEncoder_72430328479997
// MI455X (gfx1250) — matrix operand order repaired and checked by reading, not compiled, not run on hardware
//
#include <hip/hip_runtime.h>
#include <cstdint>
#include <cstddef>

#define LN_EPS 1e-5f

typedef __attribute__((ext_vector_type(16))) __bf16 bf16x16;
typedef __attribute__((ext_vector_type(8)))  float  f32x8;

// ---------- helpers ----------
static __device__ __forceinline__ unsigned short f32_to_bf16(float f) {
  unsigned u = __float_as_uint(f);
  u += 0x7FFFu + ((u >> 16) & 1u);           // round-to-nearest-even
  return (unsigned short)(u >> 16);
}
// order-preserving float -> uint mapping (for atomicMax on floats incl. negatives)
static __device__ __forceinline__ unsigned fenc(float f) {
  unsigned u = __float_as_uint(f);
  return (u & 0x80000000u) ? ~u : (u | 0x80000000u);
}
static __device__ __forceinline__ float fdec(unsigned u) {
  u = (u & 0x80000000u) ? (u & 0x7FFFFFFFu) : ~u;
  return __uint_as_float(u);
}

// ---------- small utility kernels ----------
__global__ void k_zero_f32(float* p, long long n) {
  long long i = (long long)blockIdx.x * blockDim.x + threadIdx.x;
  if (i < n) p[i] = 0.f;
}
__global__ void k_zero_u16(unsigned short* p, long long n) {
  long long i = (long long)blockIdx.x * blockDim.x + threadIdx.x;
  if (i < n) p[i] = 0;
}

// normalize skeleton inputs -> bf16 A buffer (rows [0,Npad), cols [0,32), lda=256)
__global__ void k_build_x0(const float* __restrict__ lo, const float* __restrict__ go,
                           const float* __restrict__ lom, const float* __restrict__ los,
                           const float* __restrict__ gom, const float* __restrict__ gos,
                           unsigned short* __restrict__ xb, int N, int Npad) {
  long long i = (long long)blockIdx.x * blockDim.x + threadIdx.x;
  long long total = (long long)Npad * 32;
  if (i >= total) return;
  int r = (int)(i >> 5), c = (int)(i & 31);
  float v = 0.f;
  if (r < N && c < 6) {
    if (c < 3) v = (lo[r * 3 + c] - lom[c]) / los[c];
    else       v = (go[r * 3 + (c - 3)] - gom[c - 3]) / gos[c - 3];
  }
  xb[(size_t)r * 256 + c] = f32_to_bf16(v);
}

// W (f32, [Fin x Nc] row-major) -> bf16 [Kp x Nc], zero-padded K
__global__ void k_pack_w(const float* __restrict__ W, unsigned short* __restrict__ wb,
                         int Fin, int Kp, int Nc) {
  long long i = (long long)blockIdx.x * blockDim.x + threadIdx.x;
  if (i >= (long long)Kp * Nc) return;
  int k = (int)(i / Nc), n = (int)(i % Nc);
  wb[i] = (k < Fin) ? f32_to_bf16(W[(size_t)k * Nc + n]) : (unsigned short)0;
}

// ---------- WMMA GEMM: C[M x Nc] = A[M x K](bf16, lda=256) @ B[K x Nc](bf16) ----------
// block = 256 threads (8 waves), tile = 128 rows x 64 cols, K step = 32.
// A tile staged with GLOBAL_LOAD_ASYNC_TO_LDS_B128 (ASYNCcnt), LDS double-buffered.
__global__ __launch_bounds__(256) void k_gemm_bf16_wmma(
    const unsigned short* __restrict__ A,
    const unsigned short* __restrict__ B,
    float* __restrict__ C,
    int K, int Nc) {
  __shared__ unsigned short lds_a[2][128 * 32];   // [buf][row][k]
  __shared__ unsigned short lds_b[2][64 * 32];    // [buf][n][k] (transposed)

  const int tid  = threadIdx.x;
  const int lane = tid & 31;
  const int wv   = tid >> 5;
  const int row0 = blockIdx.x * 128;
  const int n0   = blockIdx.y * 64;
  const int lda  = 256;

  const int grp = lane >> 4;     // half-wave group per ISA 16-bit layouts
  const int lm  = lane & 15;

  // stage one K-slab (32 wide) into LDS buffer `buf`
  auto stage = [&](int buf, int kb) {
    // --- A tile: 128 rows x 32 k (bf16) via async DMA to LDS (16B per lane) ---
    {
      int idx = tid;
      #pragma unroll
      for (int it = 0; it < 2; ++it, idx += 256) {
        int r = idx >> 2, seg = idx & 3;
        unsigned loff = (unsigned)(uintptr_t)&lds_a[buf][r * 32 + seg * 8];
        unsigned long long ga =
            (unsigned long long)(uintptr_t)(A + (size_t)(row0 + r) * lda + kb + seg * 8);
        asm volatile("global_load_async_to_lds_b128 %0, %1, off"
                     :: "v"(loff), "v"(ga) : "memory");
      }
    }
    // --- B tile: read rows coalesced, scatter transposed to [n][k] ---
    {
      int k = tid >> 3, nseg = tid & 7;
      const uint4 v = *(const uint4*)(B + (size_t)(kb + k) * Nc + n0 + nseg * 8);
      unsigned short e[8];
      e[0] = (unsigned short)(v.x);  e[1] = (unsigned short)(v.x >> 16);
      e[2] = (unsigned short)(v.y);  e[3] = (unsigned short)(v.y >> 16);
      e[4] = (unsigned short)(v.z);  e[5] = (unsigned short)(v.z >> 16);
      e[6] = (unsigned short)(v.w);  e[7] = (unsigned short)(v.w >> 16);
      #pragma unroll
      for (int j = 0; j < 8; ++j)
        lds_b[buf][(nseg * 8 + j) * 32 + k] = e[j];
      if (kb + 32 < K)
        __builtin_prefetch(B + (size_t)(kb + 32 + k) * Nc + n0 + nseg * 8, 0, 1);
    }
  };

  const f32x8 vzero = {0.f, 0.f, 0.f, 0.f, 0.f, 0.f, 0.f, 0.f};
  f32x8 acc[4];
  acc[0] = vzero; acc[1] = vzero; acc[2] = vzero; acc[3] = vzero;

  stage(0, 0);
  int buf = 0;
  for (int kb = 0; kb < K; kb += 32) {
    asm volatile("s_wait_asynccnt 0x0" ::: "memory");  // A-tile DMA complete (this wave)
    __syncthreads();                                   // all waves' A/B staging visible
    if (kb + 32 < K) stage(buf ^ 1, kb + 32);          // overlap next slab's DMA

    // --- A fragment: 16-bit A 16x32 layout (grp0: k 0..7,16..23; grp1: k 8..15,24..31)
    bf16x16 afrag;
    {
      const unsigned short* ap = &lds_a[buf][(wv * 16 + lm) * 32];
      ((uint4*)&afrag)[0] = *(const uint4*)(ap + 8 * grp);
      ((uint4*)&afrag)[1] = *(const uint4*)(ap + 16 + 8 * grp);
    }
    // --- preload all 4 B fragments, then issue WMMAs back-to-back ---
    bf16x16 bfr[4];
    #pragma unroll
    for (int nt = 0; nt < 4; ++nt) {
      const unsigned short* bp = &lds_b[buf][(nt * 16 + lm) * 32 + 8 * grp];
      ((uint4*)&bfr[nt])[0] = *(const uint4*)(bp);
      ((uint4*)&bfr[nt])[1] = *(const uint4*)(bp + 16);
    }
    #pragma unroll
    for (int nt = 0; nt < 4; ++nt)
      acc[nt] = __builtin_amdgcn_wmma_f32_16x16x32_bf16(
          false, afrag, false, bfr[nt], (short)0, acc[nt], false, false);

    buf ^= 1;
    __syncthreads();   // fragment reads done before next iteration overwrites this buffer
  }

  // --- store: 32-bit C/D layout, VGPR r -> M = r + 8*grp, N = lane&15 ---
  #pragma unroll
  for (int nt = 0; nt < 4; ++nt) {
    const int col = n0 + nt * 16 + lm;
    #pragma unroll
    for (int r = 0; r < 8; ++r) {
      const int row = row0 + wv * 16 + r + 8 * grp;
      C[(size_t)row * Nc + col] = acc[nt][r];
    }
  }
}

// ---------- attention ----------
__global__ void k_attn_dots(const float* __restrict__ z,
                            const float* __restrict__ avs, const float* __restrict__ avd,
                            float* __restrict__ asrc, float* __restrict__ adst,
                            int N, int H, int D) {
  long long i = (long long)blockIdx.x * blockDim.x + threadIdx.x;
  if (i >= (long long)N * H) return;
  int n = (int)(i / H), h = (int)(i % H);
  const float4* zp = (const float4*)(z + (size_t)n * D + h * 64);
  const float4* as = (const float4*)(avs + h * 64);
  const float4* ad = (const float4*)(avd + h * 64);
  float s = 0.f, d = 0.f;
  #pragma unroll
  for (int j = 0; j < 16; ++j) {
    float4 zv = zp[j], a = as[j], b = ad[j];
    s += zv.x * a.x + zv.y * a.y + zv.z * a.z + zv.w * a.w;
    d += zv.x * b.x + zv.y * b.y + zv.z * b.z + zv.w * b.w;
  }
  asrc[i] = s;
  adst[i] = d;
}

__global__ void k_init_node(unsigned* __restrict__ amax, float* __restrict__ den, int n) {
  int i = blockIdx.x * blockDim.x + threadIdx.x;
  if (i < n) { amax[i] = 0x007FFFFFu /* enc(-inf) */; den[i] = 0.f; }
}

static __device__ __forceinline__ void edge_sd(const int* ei, int E, int e, int& s, int& d) {
  if (e < E) { s = ei[e]; d = ei[E + e]; } else { s = d = e - E; }
}

__global__ void k_edge_max(const int* __restrict__ ei, int E, int Etot,
                           const float* __restrict__ asrc, const float* __restrict__ adst,
                           unsigned* __restrict__ amax, int H) {
  int e = blockIdx.x * blockDim.x + threadIdx.x;
  if (e >= Etot) return;
  int s, d; edge_sd(ei, E, e, s, d);
  for (int h = 0; h < H; ++h) {
    float a = asrc[s * H + h] + adst[d * H + h];
    a = (a > 0.f) ? a : 0.2f * a;              // leaky_relu(0.2)
    atomicMax(amax + d * H + h, fenc(a));
  }
}

__global__ void k_edge_expsum(const int* __restrict__ ei, int E, int Etot,
                              const float* __restrict__ asrc, const float* __restrict__ adst,
                              const unsigned* __restrict__ amax,
                              float* __restrict__ ex, float* __restrict__ den, int H) {
  int e = blockIdx.x * blockDim.x + threadIdx.x;
  if (e >= Etot) return;
  int s, d; edge_sd(ei, E, e, s, d);
  for (int h = 0; h < H; ++h) {
    float a = asrc[s * H + h] + adst[d * H + h];
    a = (a > 0.f) ? a : 0.2f * a;
    float m = fdec(amax[d * H + h]);
    float ev = __expf(a - m);
    ex[(size_t)e * H + h] = ev;
    atomicAdd(den + d * H + h, ev);
  }
}

// one block per edge, thread t covers feature t (h = t/64)
__global__ void k_edge_aggr(const int* __restrict__ ei, int E,
                            const float* __restrict__ z,
                            const float* __restrict__ ex, const float* __restrict__ den,
                            float* __restrict__ outb, int H, int D) {
  int e = blockIdx.x;
  int t = threadIdx.x;
  int s, d; edge_sd(ei, E, e, s, d);
  int h = t >> 6;
  float att = ex[(size_t)e * H + h] / (den[d * H + h] + 1e-16f);
  atomicAdd(outb + (size_t)d * D + t, z[(size_t)s * D + t] * att);
}

// fused bias + LayerNorm (+ ReLU) ; writes f32 final output and/or next-layer bf16 A
__global__ void k_bias_ln(const float* __restrict__ outb,
                          const float* __restrict__ bias,
                          const float* __restrict__ g, const float* __restrict__ b,
                          float* __restrict__ fo, unsigned short* __restrict__ xb,
                          int N, int D, int relu) {
  int lane = threadIdx.x & 31;
  int wid  = threadIdx.x >> 5;
  int n = blockIdx.x * (blockDim.x >> 5) + wid;
  if (n >= N) return;
  const float* row = outb + (size_t)n * D;
  const int per = D / 32;
  float vals[8];
  float s = 0.f, ss = 0.f;
  for (int i = 0; i < per; ++i) {
    int c = lane + 32 * i;
    float v = row[c] + bias[c];
    vals[i] = v; s += v; ss += v * v;
  }
  #pragma unroll
  for (int off = 16; off > 0; off >>= 1) {
    s  += __shfl_xor(s,  off, 32);
    ss += __shfl_xor(ss, off, 32);
  }
  float mu  = s / (float)D;
  float var = ss / (float)D - mu * mu;
  float rs  = rsqrtf(var + LN_EPS);
  for (int i = 0; i < per; ++i) {
    int c = lane + 32 * i;
    float y = (vals[i] - mu) * rs * g[c] + b[c];
    if (relu) y = fmaxf(y, 0.f);
    if (fo) fo[(size_t)n * D + c] = y;
    if (xb) xb[(size_t)n * 256 + c] = f32_to_bf16(y);
  }
}

// ---------- launch ----------
extern "C" void kernel_launch(void* const* d_in, const int* in_sizes, int n_in,
                              void* d_out, int out_size, void* d_ws, size_t ws_size,
                              hipStream_t stream) {
  const float* lo  = (const float*)d_in[0];
  const float* go  = (const float*)d_in[1];
  const float* lom = (const float*)d_in[2];
  const float* los = (const float*)d_in[3];
  const float* gom = (const float*)d_in[4];
  const float* gos = (const float*)d_in[5];
  const int*   ei  = (const int*)d_in[6];

  const int N    = in_sizes[0] / 3;
  const int E    = in_sizes[6] / 2;
  const int Npad = (N + 127) & ~127;
  const int Etot = E + N;

  const int FinA[4] = {6, 256, 256, 256};
  const int KpA[4]  = {32, 256, 256, 256};
  const int HA[4]   = {4, 4, 4, 1};
  const int DA[4]   = {256, 256, 256, 64};

  const float *W[4], *avs[4], *avd[4], *bi[4], *lng[4], *lnb[4];
  for (int l = 0; l < 4; ++l) {
    int b0 = 7 + 6 * l;
    W[l]   = (const float*)d_in[b0 + 0];
    avs[l] = (const float*)d_in[b0 + 1];
    avd[l] = (const float*)d_in[b0 + 2];
    bi[l]  = (const float*)d_in[b0 + 3];
    lng[l] = (const float*)d_in[b0 + 4];
    lnb[l] = (const float*)d_in[b0 + 5];
  }

  char* wp = (char*)d_ws;
  auto carve = [&](size_t bytes) -> char* {
    char* p = wp; wp += (bytes + 255) & ~(size_t)255; return p;
  };
  unsigned short* xb   = (unsigned short*)carve((size_t)Npad * 256 * 2);
  unsigned short* wb   = (unsigned short*)carve((size_t)256 * 256 * 2);
  float*          z    = (float*)carve((size_t)Npad * 256 * 4);
  float*          outb = (float*)carve((size_t)Npad * 256 * 4);
  float*          asrc = (float*)carve((size_t)N * 4 * 4);
  float*          adst = (float*)carve((size_t)N * 4 * 4);
  unsigned*       amax = (unsigned*)carve((size_t)N * 4 * 4);
  float*          den  = (float*)carve((size_t)N * 4 * 4);
  float*          ex   = (float*)carve((size_t)Etot * 4 * 4);

  {
    long long tot = (long long)Npad * 32;
    k_build_x0<<<(unsigned)((tot + 255) / 256), 256, 0, stream>>>(
        lo, go, lom, los, gom, gos, xb, N, Npad);
  }
  {
    long long tot = (long long)(Npad - N) * 256;  // zero padding rows for cols 0..255
    if (tot > 0)
      k_zero_u16<<<(unsigned)((tot + 255) / 256), 256, 0, stream>>>(xb + (size_t)N * 256, tot);
  }

  for (int l = 0; l < 4; ++l) {
    const int Fin = FinA[l], Kp = KpA[l], H = HA[l], D = DA[l];

    {
      long long tot = (long long)Kp * D;
      k_pack_w<<<(unsigned)((tot + 255) / 256), 256, 0, stream>>>(W[l], wb, Fin, Kp, D);
    }
    {
      dim3 grid(Npad / 128, D / 64);
      k_gemm_bf16_wmma<<<grid, 256, 0, stream>>>(xb, wb, z, Kp, D);
    }
    {
      long long tot = (long long)N * H;
      k_attn_dots<<<(unsigned)((tot + 255) / 256), 256, 0, stream>>>(z, avs[l], avd[l], asrc, adst, N, H, D);
      k_init_node<<<(unsigned)((tot + 255) / 256), 256, 0, stream>>>(amax, den, (int)tot);
    }
    k_edge_max<<<(Etot + 255) / 256, 256, 0, stream>>>(ei, E, Etot, asrc, adst, amax, H);
    k_edge_expsum<<<(Etot + 255) / 256, 256, 0, stream>>>(ei, E, Etot, asrc, adst, amax, ex, den, H);
    {
      long long tot = (long long)N * D;
      k_zero_f32<<<(unsigned)((tot + 255) / 256), 256, 0, stream>>>(outb, tot);
    }
    k_edge_aggr<<<Etot, D, 0, stream>>>(ei, E, z, ex, den, outb, H, D);

    const int last = (l == 3);
    k_bias_ln<<<(N + 3) / 4, 128, 0, stream>>>(
        outb, bi[l], lng[l], lnb[l],
        last ? (float*)d_out : nullptr,
        last ? nullptr : xb,
        N, D, last ? 0 : 1);
  }
}
